// MultiheadSelfAttn_43980465111306
// MI455X (gfx1250) — matrix operand order repaired and checked by reading, not compiled, not run on hardware
//
#include <hip/hip_runtime.h>
#include <hip/hip_bf16.h>

typedef __attribute__((ext_vector_type(16))) __bf16 v16bf;
typedef __attribute__((ext_vector_type(8)))  float  v8f;

#define NB 4
#define LSEQ 2048
#define DMODEL 768
#define NHEADS 12
#define HDIM 64
#define BL 8192   // NB*LSEQ

union BV {
    v16bf v;
    unsigned int u[8];
    unsigned short s[16];
};

// A-matrix (16x32 bf16) layout: lane m = L%16, half = L/16
//   VGPR i (u32 = K pair): kbase = 16*(i>>2) + 8*half + 2*(i&3)
__device__ __forceinline__ int a_kbase(int i, int half) {
    return ((i >> 2) << 4) + (half << 3) + ((i & 3) << 1);
}
// B-matrix (32x16 bf16) layout: lane n = L%16, half = L/16
//   VGPR i: kbase = 16*(i>>2) + 8*half + 2*(i&3)
__device__ __forceinline__ int b_kbase(int i, int half) {
    return ((i >> 2) << 4) + (half << 3) + ((i & 3) << 1);
}

__device__ __forceinline__ v8f wmma_bf16(const BV& a, const BV& b, v8f c) {
    return __builtin_amdgcn_wmma_f32_16x16x32_bf16(false, a.v, false, b.v,
                                                   (short)0, c, false, false);
}

// CDNA5 async global->LDS copy (ASYNCcnt tracked), per ISA 15.18.3 op 98.
__device__ __forceinline__ void async_load_b128(unsigned lds_off, const void* gaddr) {
    asm volatile("global_load_async_to_lds_b128 %0, %1, off"
                 :: "v"(lds_off), "v"(gaddr) : "memory");
}
__device__ __forceinline__ void wait_asynccnt0() {
    asm volatile("s_wait_asynccnt 0x0" ::: "memory");
}

// ---------------------------------------------------------------- f32 -> bf16
__global__ void f2bf_kernel(const float* __restrict__ src, __bf16* __restrict__ dst, int n) {
    int i = blockIdx.x * 256 + threadIdx.x;
    if (i < n) dst[i] = (__bf16)src[i];
}

// ------------------------------------------------- NT GEMM: C = A[M,K] @ B[N,K]^T
// 4 waves/block; each wave owns a 32x64 C tile (block: 128x64).
// B 64x32 tile double-buffered in LDS via async global->LDS loads.
__global__ __launch_bounds__(128)
void gemm_nt_bf16(const __bf16* __restrict__ A, const __bf16* __restrict__ Bm,
                  float* __restrict__ C, const float* __restrict__ bias,
                  const float* __restrict__ resid, int M, int N, int K)
{
    __shared__ __align__(16) __bf16 lbs[2][64 * 32];
    const int lane  = threadIdx.x & 31;
    const int wave  = threadIdx.x >> 5;
    const int half  = lane >> 4;
    const int c16   = lane & 15;
    const int mbase = blockIdx.x * 128 + wave * 32;
    const int nbase = blockIdx.y * 64;

    // staging assignment: thread t copies 32B of row nr, half hb
    const int nr = threadIdx.x >> 1, hb = threadIdx.x & 1;
    const unsigned lds0 = (unsigned)(unsigned long long)(&lbs[0][nr * 32 + hb * 16]);
    const unsigned lds1 = (unsigned)(unsigned long long)(&lbs[1][nr * 32 + hb * 16]);
    const __bf16* bsrc  = Bm + (size_t)(nbase + nr) * K + hb * 16;

    // prologue: stage k-tile 0 into buffer 0
    async_load_b128(lds0, bsrc);
    async_load_b128(lds0 + 16, bsrc + 8);

    v8f acc[2][4] = {};
    const int nk = K / 32;
    for (int ki = 0; ki < nk; ki++) {
        const int k0 = ki * 32;
        wait_asynccnt0();
        __syncthreads();
        if (ki + 1 < nk) {   // stage next tile into the other buffer
            unsigned ldsn = ((ki + 1) & 1) ? lds1 : lds0;
            const __bf16* bn = bsrc + (size_t)(k0 + 32);
            async_load_b128(ldsn, bn);
            async_load_b128(ldsn + 16, bn + 8);
        }
        const __bf16* lcur = lbs[ki & 1];
        BV a[2];
#pragma unroll
        for (int mt = 0; mt < 2; mt++)
#pragma unroll
            for (int i = 0; i < 8; i++)
                a[mt].u[i] = *(const unsigned int*)(A + (size_t)(mbase + mt * 16 + c16) * K +
                                                    k0 + a_kbase(i, half));
#pragma unroll
        for (int nt = 0; nt < 4; nt++) {
            BV b;
#pragma unroll
            for (int i = 0; i < 8; i++)
                b.u[i] = *(const unsigned int*)(&lcur[(nt * 16 + c16) * 32 + b_kbase(i, half)]);
            acc[0][nt] = wmma_bf16(a[0], b, acc[0][nt]);
            acc[1][nt] = wmma_bf16(a[1], b, acc[1][nt]);
        }
    }
#pragma unroll
    for (int mt = 0; mt < 2; mt++)
#pragma unroll
        for (int nt = 0; nt < 4; nt++)
#pragma unroll
            for (int r = 0; r < 8; r++) {
                int m  = mbase + mt * 16 + r + half * 8;
                int nn = nbase + nt * 16 + c16;
                float v = acc[mt][nt][r];
                if (bias)  v += bias[nn];
                if (resid) v += resid[(size_t)m * N + nn];
                C[(size_t)m * N + nn] = v;
            }
}

// --------------------------------- LayerNorm (no bias LN) + head split to bf16
__global__ __launch_bounds__(256)
void ln_split_kernel(const float* __restrict__ qraw, const float* __restrict__ kraw,
                     const float* __restrict__ bk, const float* __restrict__ gq,
                     const float* __restrict__ gk, __bf16* __restrict__ q_t,
                     __bf16* __restrict__ k_t)
{
    const int row   = blockIdx.x;
    const int which = blockIdx.y;
    const float* src = which ? kraw : qraw;
    const float* g   = which ? gk : gq;
    __bf16* dst      = which ? k_t : q_t;
    const float osc  = which ? 1.0f : 0.125f;   // hd^-0.5 folded into q

    __shared__ float red[256];
    float vals[3], s = 0.f, s2 = 0.f;
#pragma unroll
    for (int j = 0; j < 3; j++) {
        int d = threadIdx.x + j * 256;
        float v = src[(size_t)row * DMODEL + d];
        if (which) v += bk[d];
        vals[j] = v; s += v; s2 += v * v;
    }
    red[threadIdx.x] = s; __syncthreads();
    for (int o = 128; o > 0; o >>= 1) { if (threadIdx.x < o) red[threadIdx.x] += red[threadIdx.x + o]; __syncthreads(); }
    float mean = red[0] * (1.f / DMODEL); __syncthreads();
    red[threadIdx.x] = s2; __syncthreads();
    for (int o = 128; o > 0; o >>= 1) { if (threadIdx.x < o) red[threadIdx.x] += red[threadIdx.x + o]; __syncthreads(); }
    float var = red[0] * (1.f / DMODEL) - mean * mean;
    float rs  = rsqrtf(var + 1e-5f);
#pragma unroll
    for (int j = 0; j < 3; j++) {
        int d  = threadIdx.x + j * 256;
        int nh = d >> 6, hd = d & 63;
        float o = (vals[j] - mean) * rs * g[d] * osc;
        dst[((size_t)nh * BL + row) * HDIM + hd] = (__bf16)o;
    }
}

// ------------------------------------------------------ V bias + head split
__global__ void vbias_kernel(const float* __restrict__ vraw, const float* __restrict__ bv,
                             __bf16* __restrict__ v_t)
{
    int idx = blockIdx.x * 256 + threadIdx.x;
    if (idx >= BL * DMODEL) return;
    int row = idx / DMODEL, d = idx % DMODEL;
    int nh = d >> 6, hd = d & 63;
    v_t[((size_t)nh * BL + row) * HDIM + hd] = (__bf16)(vraw[idx] + bv[d]);
}

// ------------------- Pass 1: per-(b,n,kcol) column max & 1/sum(exp) over all q
// blockIdx.x = bn, blockIdx.y = ktile (32 columns); one wave per block.
__global__ __launch_bounds__(32)
void attn_stats_kernel(const __bf16* __restrict__ q_t, const __bf16* __restrict__ k_t,
                       float* __restrict__ mbuf, float* __restrict__ zbuf)
{
    const int bn = blockIdx.x;
    const int b  = bn & 3, n = bn >> 2;
    const size_t base = ((size_t)n * BL + (size_t)b * LSEQ) * HDIM;
    const __bf16* qp = q_t + base;
    const __bf16* kp = k_t + base;
    const int lane = threadIdx.x & 31;
    const int half = lane >> 4, c16 = lane & 15;
    const int kbase0 = blockIdx.y * 32;

    BV bk0[2], bk1[2];
#pragma unroll
    for (int kt = 0; kt < 2; kt++) {
        const int kcol = kbase0 + kt * 16 + c16;
#pragma unroll
        for (int i = 0; i < 8; i++) {
            bk0[kt].u[i] = *(const unsigned int*)(kp + (size_t)kcol * HDIM + b_kbase(i, half));
            bk1[kt].u[i] = *(const unsigned int*)(kp + (size_t)kcol * HDIM + 32 + b_kbase(i, half));
        }
    }
    float m[2] = {-3.0e38f, -3.0e38f}, z[2] = {0.f, 0.f};
    for (int q0 = 0; q0 < LSEQ; q0 += 16) {
        BV a0, a1;
#pragma unroll
        for (int i = 0; i < 8; i++) {
            a0.u[i] = *(const unsigned int*)(qp + (size_t)(q0 + c16) * HDIM + a_kbase(i, half));
            a1.u[i] = *(const unsigned int*)(qp + (size_t)(q0 + c16) * HDIM + 32 + a_kbase(i, half));
        }
#pragma unroll
        for (int kt = 0; kt < 2; kt++) {
            v8f s = {};
            s = wmma_bf16(a0, bk0[kt], s);
            s = wmma_bf16(a1, bk1[kt], s);
            float tmax = s[0];
#pragma unroll
            for (int r = 1; r < 8; r++) tmax = fmaxf(tmax, s[r]);
            float nm = fmaxf(m[kt], tmax);
            float zs = 0.f;
#pragma unroll
            for (int r = 0; r < 8; r++) zs += __expf(s[r] - nm);
            z[kt] = z[kt] * __expf(m[kt] - nm) + zs;
            m[kt] = nm;
        }
    }
#pragma unroll
    for (int kt = 0; kt < 2; kt++) {
        float om = __shfl_xor(m[kt], 16, 32);
        float oz = __shfl_xor(z[kt], 16, 32);
        float Mt = fmaxf(m[kt], om);
        float Zt = z[kt] * __expf(m[kt] - Mt) + oz * __expf(om - Mt);
        if (half == 0) {
            int idx = bn * LSEQ + kbase0 + kt * 16 + c16;
            mbuf[idx] = Mt;
            zbuf[idx] = 1.f / Zt;
        }
    }
}

// ---------------- Pass 2: o = P @ V, P = exp(s - m_k) / Z_k (column softmax)
// blockIdx.x = bn, blockIdx.y = qtile (32 rows); one wave; P routed through
// LDS to convert C-layout -> A-layout for the second WMMA chain.
__global__ __launch_bounds__(32)
void attn_av_kernel(const __bf16* __restrict__ q_t, const __bf16* __restrict__ k_t,
                    const __bf16* __restrict__ v_t, const float* __restrict__ mbuf,
                    const float* __restrict__ zbuf, __bf16* __restrict__ o_bf)
{
    const int bn = blockIdx.x;
    const int b  = bn & 3, n = bn >> 2;
    const size_t base = ((size_t)n * BL + (size_t)b * LSEQ) * HDIM;
    const __bf16* qp = q_t + base;
    const __bf16* kp = k_t + base;
    const __bf16* vp = v_t + base;
    const int lane = threadIdx.x & 31;
    const int half = lane >> 4, c16 = lane & 15;
    const int qbase = blockIdx.y * 32;

    __shared__ __align__(4) __bf16 pt[32 * 32];   // P tile [q][k]

    BV aq0[2], aq1[2];
#pragma unroll
    for (int qt = 0; qt < 2; qt++)
#pragma unroll
        for (int i = 0; i < 8; i++) {
            const size_t qrow = (size_t)(qbase + qt * 16 + c16) * HDIM;
            aq0[qt].u[i] = *(const unsigned int*)(qp + qrow + a_kbase(i, half));
            aq1[qt].u[i] = *(const unsigned int*)(qp + qrow + 32 + a_kbase(i, half));
        }
    v8f oacc[2][4] = {};
    for (int k0 = 0; k0 < LSEQ; k0 += 32) {
#pragma unroll
        for (int kt = 0; kt < 2; kt++) {
            const int kcol = k0 + kt * 16 + c16;
            BV b0, b1;
#pragma unroll
            for (int i = 0; i < 8; i++) {
                b0.u[i] = *(const unsigned int*)(kp + (size_t)kcol * HDIM + b_kbase(i, half));
                b1.u[i] = *(const unsigned int*)(kp + (size_t)kcol * HDIM + 32 + b_kbase(i, half));
            }
            float mv = mbuf[bn * LSEQ + kcol];
            float rz = zbuf[bn * LSEQ + kcol];
#pragma unroll
            for (int qt = 0; qt < 2; qt++) {
                v8f s = {};
                s = wmma_bf16(aq0[qt], b0, s);
                s = wmma_bf16(aq1[qt], b1, s);
#pragma unroll
                for (int r = 0; r < 8; r++) {
                    float p = __expf(s[r] - mv) * rz;
                    pt[(qt * 16 + r + half * 8) * 32 + kt * 16 + c16] = (__bf16)p;
                }
            }
        }
        __syncthreads();
        BV a2[2];
#pragma unroll
        for (int qt = 0; qt < 2; qt++)
#pragma unroll
            for (int i = 0; i < 8; i++)
                a2[qt].u[i] = *(const unsigned int*)(&pt[(qt * 16 + c16) * 32 + a_kbase(i, half)]);
#pragma unroll
        for (int dt = 0; dt < 4; dt++) {
            BV b2;
#pragma unroll
            for (int i = 0; i < 8; i++) {
                int kk = k0 + b_kbase(i, half);
                int dc = dt * 16 + c16;
                unsigned int lo = *(const unsigned short*)(vp + (size_t)kk * HDIM + dc);
                unsigned int hi = *(const unsigned short*)(vp + (size_t)(kk + 1) * HDIM + dc);
                b2.u[i] = lo | (hi << 16);
            }
            oacc[0][dt] = wmma_bf16(a2[0], b2, oacc[0][dt]);
            oacc[1][dt] = wmma_bf16(a2[1], b2, oacc[1][dt]);
        }
        __syncthreads();
    }
    // store o tile as bf16 in [b*l, 768] layout (head-concat) for the Wo GEMM
#pragma unroll
    for (int qt = 0; qt < 2; qt++)
#pragma unroll
        for (int dt = 0; dt < 4; dt++)
#pragma unroll
            for (int r = 0; r < 8; r++) {
                int row = b * LSEQ + qbase + qt * 16 + r + half * 8;
                int col = n * HDIM + dt * 16 + c16;
                o_bf[(size_t)row * DMODEL + col] = (__bf16)oacc[qt][dt][r];
            }
}

// ---------------------------------------------------------------------------
extern "C" void kernel_launch(void* const* d_in, const int* in_sizes, int n_in,
                              void* d_out, int out_size, void* d_ws, size_t ws_size,
                              hipStream_t stream)
{
    const float* x  = (const float*)d_in[0];
    const float* Wq = (const float*)d_in[1];
    const float* Wk = (const float*)d_in[2];
    const float* bk = (const float*)d_in[3];
    const float* Wv = (const float*)d_in[4];
    const float* bv = (const float*)d_in[5];
    const float* gq = (const float*)d_in[6];
    const float* gk = (const float*)d_in[7];
    const float* Wo = (const float*)d_in[8];
    const float* bo = (const float*)d_in[9];
    float* out = (float*)d_out;

    const size_t nXD = (size_t)BL * DMODEL;     // 6,291,456
    const size_t nW  = (size_t)DMODEL * DMODEL; // 589,824
    const size_t nH  = (size_t)NHEADS * BL * HDIM;

    char* p = (char*)d_ws;
    auto alloc = [&](size_t bytes) -> void* {
        void* r = (void*)p;
        p += (bytes + 255) & ~(size_t)255;
        return r;
    };
    __bf16* x_bf  = (__bf16*)alloc(nXD * 2);
    __bf16* wq_bf = (__bf16*)alloc(nW * 2);
    __bf16* wk_bf = (__bf16*)alloc(nW * 2);
    __bf16* wv_bf = (__bf16*)alloc(nW * 2);
    __bf16* wo_bf = (__bf16*)alloc(nW * 2);
    float*  qraw  = (float*)alloc(nXD * 4);
    float*  kraw  = (float*)alloc(nXD * 4);
    float*  vraw  = (float*)alloc(nXD * 4);
    __bf16* q_t   = (__bf16*)alloc(nH * 2);
    __bf16* k_t   = (__bf16*)alloc(nH * 2);
    __bf16* v_t   = (__bf16*)alloc(nH * 2);
    float*  mbuf  = (float*)alloc((size_t)NB * NHEADS * LSEQ * 4);
    float*  zbuf  = (float*)alloc((size_t)NB * NHEADS * LSEQ * 4);
    __bf16* o_bf  = (__bf16*)qraw;   // reuse: qraw dead after LN

    // 1) fp32 -> bf16 conversions
    f2bf_kernel<<<dim3((nXD + 255) / 256), 256, 0, stream>>>(x, x_bf, (int)nXD);
    f2bf_kernel<<<dim3((nW + 255) / 256), 256, 0, stream>>>(Wq, wq_bf, (int)nW);
    f2bf_kernel<<<dim3((nW + 255) / 256), 256, 0, stream>>>(Wk, wk_bf, (int)nW);
    f2bf_kernel<<<dim3((nW + 255) / 256), 256, 0, stream>>>(Wv, wv_bf, (int)nW);
    f2bf_kernel<<<dim3((nW + 255) / 256), 256, 0, stream>>>(Wo, wo_bf, (int)nW);

    // 2) QKV projections: raw = x @ W^T
    dim3 ggrid(BL / 128, DMODEL / 64);
    gemm_nt_bf16<<<ggrid, 128, 0, stream>>>(x_bf, wq_bf, qraw, nullptr, nullptr, BL, DMODEL, DMODEL);
    gemm_nt_bf16<<<ggrid, 128, 0, stream>>>(x_bf, wk_bf, kraw, nullptr, nullptr, BL, DMODEL, DMODEL);
    gemm_nt_bf16<<<ggrid, 128, 0, stream>>>(x_bf, wv_bf, vraw, nullptr, nullptr, BL, DMODEL, DMODEL);

    // 3) QK layernorm + head split (q pre-scaled by hd^-0.5); V bias + split
    ln_split_kernel<<<dim3(BL, 2), 256, 0, stream>>>(qraw, kraw, bk, gq, gk, q_t, k_t);
    vbias_kernel<<<dim3((nXD + 255) / 256), 256, 0, stream>>>(vraw, bv, v_t);

    // 4) column-softmax stats (softmax over the QUERY axis)
    attn_stats_kernel<<<dim3(NB * NHEADS, LSEQ / 32), 32, 0, stream>>>(q_t, k_t, mbuf, zbuf);

    // 5) o = P @ V
    attn_av_kernel<<<dim3(NB * NHEADS, LSEQ / 32), 32, 0, stream>>>(q_t, k_t, v_t, mbuf, zbuf, o_bf);

    // 6) out = o @ Wo^T + bo + x
    gemm_nt_bf16<<<ggrid, 128, 0, stream>>>(o_bf, wo_bf, out, bo, x, BL, DMODEL, DMODEL);
}
